// DotProductAttention_45337674776965
// MI455X (gfx1250) — matrix operand order repaired and checked by reading, not compiled, not run on hardware
//
#include <hip/hip_runtime.h>

// Flash attention forward (causal), fp32 in/out, f16 WMMA compute on gfx1250.
// B=2, H=16, S=2048, D=64.
//
// Transposed-score formulation: S^T = K Q^T and O^T = V^T P^T, so each query
// is one lane-column of the WMMA C layout -> softmax reductions are in-lane,
// and P^T feeds the PV WMMA's B operand directly (no LDS transpose).

#define BATCH   2
#define HEADS   16
#define SEQ     2048
#define DHEAD   64
#define QBLK    128           // queries per workgroup (8 waves x 16 rows)
#define KBLK    128           // keys staged in LDS per iteration
#define KS_STRIDE 72          // padded f16 stride for K tile rows   (144B, 16B-aligned groups)
#define VT_STRIDE 136         // padded f16 stride for V^T tile rows (272B, 16B-aligned groups)

typedef __attribute__((ext_vector_type(16))) _Float16 v16h;
typedef __attribute__((ext_vector_type(8)))  _Float16 h8;
typedef __attribute__((ext_vector_type(8)))  float    v8f;
typedef __attribute__((ext_vector_type(8)))  int      v8i;

static __device__ __forceinline__ v16h cat16(h8 lo, h8 hi) {
  return __builtin_shufflevector(lo, hi, 0,1,2,3,4,5,6,7,8,9,10,11,12,13,14,15);
}

static __device__ __forceinline__ v8f wmma_f16(v16h a, v16h b, v8f c) {
  // D(16x16 f32) = A(16x32 f16) * B(32x16 f16) + C
  return __builtin_amdgcn_wmma_f32_16x16x32_f16(
      /*neg_a=*/false, a, /*neg_b=*/false, b,
      /*c_mod=*/(short)0, c, /*reuse_a=*/false, /*reuse_b=*/false);
}

static __device__ __forceinline__ h8 cvt8(float4 a, float4 b, float s) {
  h8 h;
  h[0] = (_Float16)(a.x * s); h[1] = (_Float16)(a.y * s);
  h[2] = (_Float16)(a.z * s); h[3] = (_Float16)(a.w * s);
  h[4] = (_Float16)(b.x * s); h[5] = (_Float16)(b.y * s);
  h[6] = (_Float16)(b.z * s); h[7] = (_Float16)(b.w * s);
  return h;
}

// Exchange with the partner lane in the other 16-lane half (wave32), as a
// single VALU op: v_permlanex16_b32 with identity lane selects.
static __device__ __forceinline__ int xor16_i(int x) {
  return __builtin_amdgcn_permlanex16(x, x, 0x76543210, (int)0xfedcba98u,
                                      false, false);
}
static __device__ __forceinline__ float xor16_f(float x) {
  return __builtin_bit_cast(float, xor16_i(__builtin_bit_cast(int, x)));
}

__global__ __launch_bounds__(256, 1)
void fa_fwd_causal_f16wmma(const float* __restrict__ Q,
                           const float* __restrict__ K,
                           const float* __restrict__ V,
                           float* __restrict__ O) {
  __shared__ _Float16 ks[KBLK * KS_STRIDE];      // K tile, [key][dim], f16
  __shared__ _Float16 vt[DHEAD * VT_STRIDE];     // V tile transposed, [dim][key], f16

  const int tid  = threadIdx.x;
  const int lane = tid & 31;
  const int r    = lane & 15;   // A-row / B-column index within 16
  const int hg   = lane >> 4;   // half-group select per ISA WMMA layouts

  // Wave id is uniform; make it visible to the compiler (scalar control flow).
  const int wave = __builtin_amdgcn_readfirstlane(tid >> 5);

  const int bh = blockIdx.y;                    // fused (b,h): 0..31
  const int q0 = blockIdx.x * QBLK;             // first query of this block
  const int qt = q0 + wave * 16;                // first query row of this wave (SGPR)
  const int q  = qt + r;                        // this lane's query row

  const size_t base = (size_t)bh * SEQ * DHEAD;
  const float* Qg = Q + base;
  const float* Kg = K + base;
  const float* Vg = V + base;
  float*       Og = O + base;

  // ---- Q^T as WMMA B operand (scale folded in).
  // B 32x16 layout: lane r = column (query), elements e -> K row 16*hg + e
  // (contiguous 16 dims per lane group).
  v16h qb[2];
  {
    const float* qrow = Qg + (size_t)q * DHEAD;
    const float s = 0.125f; // 1/sqrt(64)
#pragma unroll
    for (int c = 0; c < 2; ++c) {
      const int d0 = c * 32 + 16 * hg;
      float4 f0 = *(const float4*)(qrow + d0);
      float4 f1 = *(const float4*)(qrow + d0 + 4);
      float4 f2 = *(const float4*)(qrow + d0 + 8);
      float4 f3 = *(const float4*)(qrow + d0 + 12);
      qb[c] = cat16(cvt8(f0, f1, s), cvt8(f2, f3, s));
    }
  }

  // ---- Online softmax state: one query per lane (replicated across hg halves).
  float m_run = -1.0e30f;
  float l_run = 0.0f;
  v8f   o_acc[4];                 // O^T: reg i of tile t = dim 16t+i+8*hg, col = query r
#pragma unroll
  for (int t = 0; t < 4; ++t) { v8f z = {}; o_acc[t] = z; }

  const int kend = q0 + QBLK;     // causal: keys < q0+128 (uniform over workgroup)

  for (int kb = 0; kb < kend; kb += KBLK) {
    // ---- Cooperative stage: K -> ks (f16, [key][dim]), V -> vt (f16, [dim][key]).
    {
      const int key  = tid >> 1;        // 0..127
      const int half = tid & 1;         // dim half: 0..31 / 32..63
      const float* krow = Kg + (size_t)(kb + key) * DHEAD + half * 32;
      const float* vrow = Vg + (size_t)(kb + key) * DHEAD + half * 32;
#pragma unroll
      for (int g = 0; g < 4; ++g) {
        float4 a = *(const float4*)(krow + 8 * g);
        float4 b = *(const float4*)(krow + 8 * g + 4);
        *(h8*)&ks[key * KS_STRIDE + half * 32 + 8 * g] = cvt8(a, b, 1.0f);
      }
#pragma unroll
      for (int g = 0; g < 4; ++g) {
        float4 a = *(const float4*)(vrow + 8 * g);
        float4 b = *(const float4*)(vrow + 8 * g + 4);
        const int d0 = half * 32 + 8 * g;
        vt[(d0 + 0) * VT_STRIDE + key] = (_Float16)a.x;
        vt[(d0 + 1) * VT_STRIDE + key] = (_Float16)a.y;
        vt[(d0 + 2) * VT_STRIDE + key] = (_Float16)a.z;
        vt[(d0 + 3) * VT_STRIDE + key] = (_Float16)a.w;
        vt[(d0 + 4) * VT_STRIDE + key] = (_Float16)b.x;
        vt[(d0 + 5) * VT_STRIDE + key] = (_Float16)b.y;
        vt[(d0 + 6) * VT_STRIDE + key] = (_Float16)b.z;
        vt[(d0 + 7) * VT_STRIDE + key] = (_Float16)b.w;
      }
      if (kb + KBLK < kend) {
        __builtin_prefetch(Kg + (size_t)(kb + KBLK + key) * DHEAD, 0, 1);
        __builtin_prefetch(Vg + (size_t)(kb + KBLK + key) * DHEAD, 0, 1);
      }
    }
    __syncthreads();

    // ---- Process the staged keys in chunks of 32; trip count is scalar.
    const int kc_end0 = qt + 16 - kb;            // causal limit for this wave
    const int kc_end  = kc_end0 < KBLK ? kc_end0 : KBLK;

    for (int kc = 0; kc < kc_end; kc += 32) {
      const int kglob = kb + kc;

      // S^T = K Q^T: two 16x16 tiles; rows = keys, cols = queries.
      // A 16x32 layout: lane r = key row; elements e -> dim 16*hg + e (contiguous).
      v8f c0 = {}, c1 = {};
#pragma unroll
      for (int dch = 0; dch < 2; ++dch) {
        h8 a0lo = *(const h8*)&ks[(kc + r)      * KS_STRIDE + dch * 32 + 16 * hg];
        h8 a0hi = *(const h8*)&ks[(kc + r)      * KS_STRIDE + dch * 32 + 16 * hg + 8];
        h8 a1lo = *(const h8*)&ks[(kc + 16 + r) * KS_STRIDE + dch * 32 + 16 * hg];
        h8 a1hi = *(const h8*)&ks[(kc + 16 + r) * KS_STRIDE + dch * 32 + 16 * hg + 8];
        c0 = wmma_f16(cat16(a0lo, a0hi), qb[dch], c0);
        c1 = wmma_f16(cat16(a1lo, a1hi), qb[dch], c1);
      }

      // Causal mask near the diagonal (scalar gate; compares vs inline consts).
      // c0 reg i = key kglob+i+8*hg; c1 reg i = key kglob+16+i+8*hg; col = query q.
      if (kglob + 31 > qt) {
        const int thr = q - kglob - 8 * hg;   // mask if key offset > thr
#pragma unroll
        for (int i = 0; i < 8; ++i) {
          if (i      > thr) c0[i] = -1.0e30f;
          if (i + 16 > thr) c1[i] = -1.0e30f;
        }
      }

      // Online softmax: keys live in-lane -> VALU tree + one half-exchange.
      float mx = fmaxf(c0[0], c1[0]);
#pragma unroll
      for (int i = 1; i < 8; ++i) mx = fmaxf(mx, fmaxf(c0[i], c1[i]));
      mx = fmaxf(mx, xor16_f(mx));

      const float mn   = fmaxf(m_run, mx);
      const float corr = __expf(m_run - mn);
      m_run = mn;

      float e0[8], e1[8], sum = 0.0f;
#pragma unroll
      for (int i = 0; i < 8; ++i) {
        e0[i] = __expf(c0[i] - mn);
        e1[i] = __expf(c1[i] - mn);
        sum += e0[i] + e1[i];
      }
      sum += xor16_f(sum);
      l_run = l_run * corr + sum;
#pragma unroll
      for (int t = 0; t < 4; ++t)
#pragma unroll
        for (int i = 0; i < 8; ++i) o_acc[t][i] *= corr;

      // Build P^T as the PV B operand in registers.
      // To match A's contiguous k order, B needs lane(hg=0): keys 0..15, lane(hg=1): keys 16..31 of the
      // chunk; each lane holds keys {8hg..8hg+7} (e0) and {16+8hg..23+8hg} (e1).
      // Send the half the partner needs, exchange once, merge.
      int own0[4], own1[4];
#pragma unroll
      for (int j = 0; j < 4; ++j) {
        own0[j] = __builtin_bit_cast(int, __builtin_amdgcn_cvt_pkrtz(e0[2*j], e0[2*j+1]));
        own1[j] = __builtin_bit_cast(int, __builtin_amdgcn_cvt_pkrtz(e1[2*j], e1[2*j+1]));
      }
      v8i pw;
#pragma unroll
      for (int j = 0; j < 4; ++j) {
        const int send = hg ? own0[j] : own1[j];  // what the partner half needs
        const int recv = xor16_i(send);
        pw[j]     = hg ? recv   : own0[j];        // elements 0..7  -> keys 16hg+0..7
        pw[4 + j] = hg ? own1[j] : recv;          // elements 8..15 -> keys 16hg+8..15
      }
      const v16h pb = __builtin_bit_cast(v16h, pw);

      // O^T += V^T P^T: A = V^T tile (lane r = dim row), B = P^T (built above).
#pragma unroll
      for (int t = 0; t < 4; ++t) {
        h8 alo = *(const h8*)&vt[(t * 16 + r) * VT_STRIDE + kc + 16 * hg];
        h8 ahi = *(const h8*)&vt[(t * 16 + r) * VT_STRIDE + kc + 16 * hg + 8];
        o_acc[t] = wmma_f16(cat16(alo, ahi), pb, o_acc[t]);
      }
    }
    __syncthreads();
  }

  // ---- Normalize and write O (fp32). Lane owns query row q; reg i of tile t
  // is dim 16t+i+8*hg -> two contiguous float4 stores per tile.
  {
    const float inv = 1.0f / l_run;
    float* orow = Og + (size_t)q * DHEAD;
#pragma unroll
    for (int t = 0; t < 4; ++t) {
      float4 lo, hi;
      lo.x = o_acc[t][0] * inv; lo.y = o_acc[t][1] * inv;
      lo.z = o_acc[t][2] * inv; lo.w = o_acc[t][3] * inv;
      hi.x = o_acc[t][4] * inv; hi.y = o_acc[t][5] * inv;
      hi.z = o_acc[t][6] * inv; hi.w = o_acc[t][7] * inv;
      *(float4*)(orow + t * 16 + 8 * hg)     = lo;
      *(float4*)(orow + t * 16 + 8 * hg + 4) = hi;
    }
  }
}

extern "C" void kernel_launch(void* const* d_in, const int* in_sizes, int n_in,
                              void* d_out, int out_size, void* d_ws, size_t ws_size,
                              hipStream_t stream) {
  (void)in_sizes; (void)n_in; (void)d_ws; (void)ws_size; (void)out_size;
  const float* Q = (const float*)d_in[0];
  const float* K = (const float*)d_in[1];
  const float* V = (const float*)d_in[2];
  // d_in[3] = padding_mask (all true), d_in[4] = causal attention mask:
  // both constant per the reference setup; causality is implemented
  // structurally in the kernel.
  float* out = (float*)d_out;

  dim3 grid(SEQ / QBLK, BATCH * HEADS);
  dim3 block(256);
  fa_fwd_causal_f16wmma<<<grid, block, 0, stream>>>(Q, K, V, out);
}
